// MultiHeadAttention_4698694222110
// MI455X (gfx1250) — matrix operand order repaired and checked by reading, not compiled, not run on hardware
//
#include <hip/hip_runtime.h>

typedef _Float16 half8  __attribute__((ext_vector_type(8)));
typedef _Float16 half16 __attribute__((ext_vector_type(16)));
typedef float    f32x8  __attribute__((ext_vector_type(8)));

#define D_MODEL 1024
#define NH      16
#define DKV     64
#define BATCH   2
#define SEQ     2048
#define TOKENS  (BATCH*SEQ)      // 4096

// ---- workspace layout (bytes) ----
#define OFF_XB   0ull                                    // 4096*1024 f16 = 8 MiB
#define OFF_WQT  (OFF_XB  + (size_t)TOKENS*D_MODEL*2)
#define OFF_WKT  (OFF_WQT + (size_t)D_MODEL*D_MODEL*2)
#define OFF_WVT  (OFF_WKT + (size_t)D_MODEL*D_MODEL*2)
#define OFF_WOT  (OFF_WVT + (size_t)D_MODEL*D_MODEL*2)
#define OFF_QB   (OFF_WOT + (size_t)D_MODEL*D_MODEL*2)   // [b,h,s,64] f16
#define OFF_KB   (OFF_QB  + (size_t)TOKENS*D_MODEL*2)    // [b,h,s,64] f16
#define OFF_VT   (OFF_KB  + (size_t)TOKENS*D_MODEL*2)    // [b,h,64,s] f16
#define OFF_HID  (OFF_VT  + (size_t)TOKENS*D_MODEL*2)    // [t, h*64+d] f16
#define OFF_BIAS (OFF_HID + (size_t)TOKENS*D_MODEL*2)    // [h][4096] f32

__device__ __forceinline__ f32x8 wmma16(half16 a, half16 b, f32x8 c) {
  return __builtin_amdgcn_wmma_f32_16x16x32_f16(false, a, false, b, (short)0, c,
                                                false, false);
}

// A-fragment: 16x32 tile, rows rowBase.., cols k0.., src row-major (ld halfs).
// Lane L (L&15 = row M): holds K = halfK*8 + j (j<8) and 16 + halfK*8 + (j-8).
__device__ __forceinline__ half16 load_a(const _Float16* __restrict__ src, int ld,
                                         int rowBase, int k0, int lane) {
  int m = lane & 15, hf = lane >> 4;
  const _Float16* p = src + (size_t)(rowBase + m) * ld + k0 + hf * 8;
  half8 lo = *(const half8*)p;
  half8 hi = *(const half8*)(p + 16);
  return __builtin_shufflevector(lo, hi, 0,1,2,3,4,5,6,7,8,9,10,11,12,13,14,15);
}

// B-fragment: 32x16 tile B[k][n]; stored transposed Bt[n][k] row-major (ld halfs).
// Lane L (L&15 = col N): holds K = half*8 + j (j<8) and 16 + half*8 + (j-8).
__device__ __forceinline__ half16 load_b(const _Float16* __restrict__ bt, int ld,
                                         int colBase, int k0, int lane) {
  int n = lane & 15, hf = lane >> 4;
  const _Float16* p = bt + (size_t)(colBase + n) * ld + k0 + hf * 8;
  half8 lo = *(const half8*)p;
  half8 hi = *(const half8*)(p + 16);
  return __builtin_shufflevector(lo, hi, 0,1,2,3,4,5,6,7,8,9,10,11,12,13,14,15);
}

// -------- stage 0: convert X to f16 --------
__global__ __launch_bounds__(256)
void convert_x_kernel(const float* __restrict__ in, _Float16* __restrict__ out, int n) {
  int i = blockIdx.x * 256 + threadIdx.x;
  if (i < n) out[i] = (_Float16)in[i];
}

// -------- stage 0: transpose+convert 1024x1024 weight to f16 Wt[n][k] --------
__global__ __launch_bounds__(256)
void transpose_convert_kernel(const float* __restrict__ in, _Float16* __restrict__ out) {
  __shared__ float tile[32][33];
  int tx = threadIdx.x & 31;
  int ty = threadIdx.x >> 5;                 // 0..7
  int bx = blockIdx.x * 32;                  // input col base
  int by = blockIdx.y * 32;                  // input row base
  for (int i = 0; i < 32; i += 8)
    tile[ty + i][tx] = in[(size_t)(by + ty + i) * D_MODEL + bx + tx];
  __syncthreads();
  for (int i = 0; i < 32; i += 8)
    out[(size_t)(bx + ty + i) * D_MODEL + by + tx] = (_Float16)tile[tx][ty + i];
}

// -------- stage 0: T5 relative-bias table biasT[h][d+2047] --------
__global__ __launch_bounds__(256)
void bias_table_kernel(const float* __restrict__ rel_emb, float* __restrict__ biasT) {
  int idx = blockIdx.x * 256 + threadIdx.x;
  if (idx >= NH * 4095) return;
  int h  = idx / 4095;
  int di = idx % 4095;
  int d  = di - 2047;                        // relative position (k - q)
  int ret = (d > 0) ? 16 : 0;                // bidirectional half-bucket offset
  int n   = (d < 0) ? -d : d;
  int bucket;
  if (n < 8) bucket = ret + n;
  else {
    int v = 8 + (int)(__logf((float)n * 0.125f) * (8.0f / __logf(16.0f)));
    bucket = ret + (v < 15 ? v : 15);
  }
  biasT[h * 4096 + di] = rel_emb[bucket * NH + h];
}

// -------- stage 1: QKV projections, wave = 32x32 tile --------
__global__ __launch_bounds__(128)
void qkv_gemm_kernel(const _Float16* __restrict__ Xb,
                     const _Float16* __restrict__ WqT,
                     const _Float16* __restrict__ WkT,
                     const _Float16* __restrict__ WvT,
                     _Float16* __restrict__ Qb,
                     _Float16* __restrict__ Kb,
                     _Float16* __restrict__ Vt) {
  int lane = threadIdx.x & 31;
  int wave = threadIdx.x >> 5;
  int which = blockIdx.z;
  const _Float16* Wt = (which == 0) ? WqT : (which == 1) ? WkT : WvT;
  int nBase = blockIdx.x * 32;
  int mBase = (blockIdx.y * 4 + wave) * 32;

  f32x8 acc[2][2] = {};
  for (int k0 = 0; k0 < D_MODEL; k0 += 32) {
    half16 a0 = load_a(Xb, D_MODEL, mBase,      k0, lane);
    half16 a1 = load_a(Xb, D_MODEL, mBase + 16, k0, lane);
    half16 b0 = load_b(Wt, D_MODEL, nBase,      k0, lane);
    half16 b1 = load_b(Wt, D_MODEL, nBase + 16, k0, lane);
    acc[0][0] = wmma16(a0, b0, acc[0][0]);
    acc[0][1] = wmma16(a0, b1, acc[0][1]);
    acc[1][0] = wmma16(a1, b0, acc[1][0]);
    acc[1][1] = wmma16(a1, b1, acc[1][1]);
  }

  int n = lane & 15, halfD = lane >> 4;     // D-layout: col = n, rows r + 8*halfD
  for (int i = 0; i < 2; i++)
    for (int j = 0; j < 2; j++) {
      int col = nBase + j * 16 + n;
      int h = col >> 6, d = col & 63;
      int tBase = mBase + i * 16 + halfD * 8;
      if (which < 2) {
        _Float16* out = (which == 0) ? Qb : Kb;
        for (int r = 0; r < 8; r++) {
          int t = tBase + r;
          int b = t >> 11, s = t & 2047;
          out[(size_t)(((b * NH + h) * SEQ + s)) * DKV + d] = (_Float16)acc[i][j][r];
        }
      } else {                               // V transposed: Vt[b,h,d,s]
        int b = tBase >> 11, s = tBase & 2047;
        half8 v;
        for (int r = 0; r < 8; r++) v[r] = (_Float16)acc[i][j][r];
        *(half8*)(Vt + ((size_t)((b * NH + h) * DKV + d)) * SEQ + s) = v;
      }
    }
}

// -------- stage 2: flash attention, wave = 16 query rows --------
__global__ __launch_bounds__(128)
void attn_kernel(const _Float16* __restrict__ Qb,
                 const _Float16* __restrict__ Kb,
                 const _Float16* __restrict__ Vt,
                 const float* __restrict__ biasT,
                 _Float16* __restrict__ Hid) {
  __shared__ float biasRow[4096];
  __shared__ __align__(16) _Float16 pstage[4][16 * 32];
  int lane = threadIdx.x & 31;
  int wave = threadIdx.x >> 5;
  int bh = blockIdx.y;                       // 0..31
  int b = bh >> 4, h = bh & 15;

  for (int i = threadIdx.x; i < 4096; i += 128) biasRow[i] = biasT[h * 4096 + i];
  __syncthreads();

  int qBase = (blockIdx.x * 4 + wave) * 16;
  const _Float16* Qh = Qb + (size_t)bh * SEQ * DKV;
  const _Float16* Kh = Kb + (size_t)bh * SEQ * DKV;
  const _Float16* Vh = Vt + (size_t)bh * DKV * SEQ;
  _Float16* ps = pstage[wave];

  half16 qa0 = load_a(Qh, DKV, qBase, 0,  lane);   // d = 0..31
  half16 qa1 = load_a(Qh, DKV, qBase, 32, lane);   // d = 32..63

  float mrow[8], lrow[8];
  for (int r = 0; r < 8; r++) { mrow[r] = -1e30f; lrow[r] = 0.f; }
  f32x8 o[4] = {};

  int nD = lane & 15, halfD = lane >> 4;     // D-layout mapping

  for (int kb = 0; kb < SEQ; kb += 32) {
    // scores: two 16-key tiles, K-dim = d (64) in two chunks
    f32x8 s[2];
    for (int t = 0; t < 2; t++) {
      half16 b0 = load_b(Kh, DKV, kb + t * 16, 0,  lane);
      half16 b1 = load_b(Kh, DKV, kb + t * 16, 32, lane);
      f32x8 z = {};
      z = wmma16(qa0, b0, z);
      z = wmma16(qa1, b1, z);
      s[t] = z;
    }
    // bias + rowwise max (16-lane half reductions match D-layout rows)
    float corr[8];
    for (int r = 0; r < 8; r++) {
      int q = qBase + r + halfD * 8;
      float v0 = s[0][r] + biasRow[(kb +      nD) - q + 2047];
      float v1 = s[1][r] + biasRow[(kb + 16 + nD) - q + 2047];
      s[0][r] = v0; s[1][r] = v1;
      float mx = fmaxf(v0, v1);
      for (int off = 1; off < 16; off <<= 1) mx = fmaxf(mx, __shfl_xor(mx, off, 32));
      float mNew = fmaxf(mrow[r], mx);
      corr[r] = __expf(mrow[r] - mNew);
      mrow[r] = mNew;
    }
    // exponentiate, row-sum, stage P (f16) to LDS in [q][key] order
    for (int r = 0; r < 8; r++) {
      float p0 = __expf(s[0][r] - mrow[r]);
      float p1 = __expf(s[1][r] - mrow[r]);
      float sum = p0 + p1;
      for (int off = 1; off < 16; off <<= 1) sum += __shfl_xor(sum, off, 32);
      lrow[r] = lrow[r] * corr[r] + sum;
      int row = r + halfD * 8;
      ps[row * 32 + nD]      = (_Float16)p0;
      ps[row * 32 + 16 + nD] = (_Float16)p1;
    }
    asm volatile("s_wait_dscnt 0" ::: "memory");   // wave-internal LDS RAW
    // P as A-fragment (16x32 over keys), V as B-fragments from Vt
    {
      int m = lane & 15, ha = lane >> 4;
      const _Float16* pp = ps + m * 32 + ha * 8;
      half8 lo = *(const half8*)pp;
      half8 hi = *(const half8*)(pp + 16);
      half16 pa = __builtin_shufflevector(lo, hi, 0,1,2,3,4,5,6,7,8,9,10,11,12,13,14,15);
      for (int dt = 0; dt < 4; dt++) {
        half16 vb = load_b(Vh, SEQ, dt * 16, kb, lane);
        f32x8 oc = o[dt];
        for (int r = 0; r < 8; r++) oc[r] *= corr[r];
        o[dt] = wmma16(pa, vb, oc);
      }
    }
  }

  // normalize + write hidden (f16, [token][h*64+d])
  for (int r = 0; r < 8; r++) lrow[r] = 1.0f / lrow[r];
  for (int dt = 0; dt < 4; dt++)
    for (int r = 0; r < 8; r++) {
      int row = r + halfD * 8;
      int t = b * SEQ + qBase + row;
      int col = h * DKV + dt * 16 + nD;
      Hid[(size_t)t * D_MODEL + col] = (_Float16)(o[dt][r] * lrow[r]);
    }
}

// -------- stage 3: output projection, f32 result --------
__global__ __launch_bounds__(128)
void out_gemm_kernel(const _Float16* __restrict__ Hid,
                     const _Float16* __restrict__ WoT,
                     float* __restrict__ Out) {
  int lane = threadIdx.x & 31;
  int wave = threadIdx.x >> 5;
  int nBase = blockIdx.x * 32;
  int mBase = (blockIdx.y * 4 + wave) * 32;

  f32x8 acc[2][2] = {};
  for (int k0 = 0; k0 < D_MODEL; k0 += 32) {
    half16 a0 = load_a(Hid, D_MODEL, mBase,      k0, lane);
    half16 a1 = load_a(Hid, D_MODEL, mBase + 16, k0, lane);
    half16 b0 = load_b(WoT, D_MODEL, nBase,      k0, lane);
    half16 b1 = load_b(WoT, D_MODEL, nBase + 16, k0, lane);
    acc[0][0] = wmma16(a0, b0, acc[0][0]);
    acc[0][1] = wmma16(a0, b1, acc[0][1]);
    acc[1][0] = wmma16(a1, b0, acc[1][0]);
    acc[1][1] = wmma16(a1, b1, acc[1][1]);
  }
  int n = lane & 15, halfD = lane >> 4;
  for (int i = 0; i < 2; i++)
    for (int j = 0; j < 2; j++) {
      int col = nBase + j * 16 + n;
      int tBase = mBase + i * 16 + halfD * 8;
      for (int r = 0; r < 8; r++)
        Out[(size_t)(tBase + r) * D_MODEL + col] = acc[i][j][r];
    }
}

extern "C" void kernel_launch(void* const* d_in, const int* in_sizes, int n_in,
                              void* d_out, int out_size, void* d_ws, size_t ws_size,
                              hipStream_t stream) {
  (void)in_sizes; (void)n_in; (void)out_size; (void)ws_size;
  const float* X       = (const float*)d_in[0];
  const float* Wq      = (const float*)d_in[1];
  const float* Wk      = (const float*)d_in[2];
  const float* Wv      = (const float*)d_in[3];
  const float* Wo      = (const float*)d_in[4];
  const float* rel_emb = (const float*)d_in[5];

  char* ws = (char*)d_ws;
  _Float16* Xb   = (_Float16*)(ws + OFF_XB);
  _Float16* WqT  = (_Float16*)(ws + OFF_WQT);
  _Float16* WkT  = (_Float16*)(ws + OFF_WKT);
  _Float16* WvT  = (_Float16*)(ws + OFF_WVT);
  _Float16* WoT  = (_Float16*)(ws + OFF_WOT);
  _Float16* Qb   = (_Float16*)(ws + OFF_QB);
  _Float16* Kb   = (_Float16*)(ws + OFF_KB);
  _Float16* Vt   = (_Float16*)(ws + OFF_VT);
  _Float16* Hid  = (_Float16*)(ws + OFF_HID);
  float*    biasT = (float*)(ws + OFF_BIAS);

  // stage 0: conversions / tables
  convert_x_kernel<<<(TOKENS * D_MODEL) / 256, 256, 0, stream>>>(X, Xb, TOKENS * D_MODEL);
  dim3 tgrid(32, 32);
  transpose_convert_kernel<<<tgrid, 256, 0, stream>>>(Wq, WqT);
  transpose_convert_kernel<<<tgrid, 256, 0, stream>>>(Wk, WkT);
  transpose_convert_kernel<<<tgrid, 256, 0, stream>>>(Wv, WvT);
  transpose_convert_kernel<<<tgrid, 256, 0, stream>>>(Wo, WoT);
  bias_table_kernel<<<(NH * 4095 + 255) / 256, 256, 0, stream>>>(rel_emb, biasT);

  // stage 1: QKV projections (z = Q,K,V)
  qkv_gemm_kernel<<<dim3(D_MODEL / 32, TOKENS / 32 / 4, 3), 128, 0, stream>>>(
      Xb, WqT, WkT, WvT, Qb, Kb, Vt);

  // stage 2: flash attention (x = query-tile groups, y = b*h)
  attn_kernel<<<dim3(SEQ / 16 / 4, BATCH * NH), 128, 0, stream>>>(
      Qb, Kb, Vt, biasT, Hid);

  // stage 3: output projection
  out_gemm_kernel<<<dim3(D_MODEL / 32, TOKENS / 32 / 4), 128, 0, stream>>>(
      Hid, WoT, (float*)d_out);
}
